// MAB_51316269253041
// MI455X (gfx1250) — matrix operand order repaired and checked by reading, not compiled, not run on hardware
//
#include <hip/hip_runtime.h>
#include <math.h>

// ---------------------------------------------------------------------------
// MAB (multihead attention block) for MI455X / gfx1250, bf16 WMMA everywhere.
// B=16, N=1024, D=512, H=8, hd=64.
// Round 2: attention K/V tiles staged cooperatively into LDS via
// GLOBAL_LOAD_ASYNC_TO_LDS_B128 (ASYNCcnt) instead of per-wave global loads.
// ---------------------------------------------------------------------------

typedef __attribute__((ext_vector_type(16))) __bf16 v16bf;
typedef __attribute__((ext_vector_type(8)))  __bf16 v8bf;
typedef __attribute__((ext_vector_type(8)))  float  v8f;

#define BATCH  16
#define SEQ    1024
#define DIM    512
#define HEADS  8
#define HDIM   64
#define ACT    (BATCH * SEQ * DIM)      // 8,388,608 elements
#define LN_EPS 1e-5f

// Load two 16-byte chunks into one WMMA operand vector (16 bf16).
static __device__ inline v16bf ld_pair(const __bf16* p0, const __bf16* p1) {
    v8bf lo = *reinterpret_cast<const v8bf*>(p0);
    v8bf hi = *reinterpret_cast<const v8bf*>(p1);
    v16bf r;
#pragma unroll
    for (int i = 0; i < 8; ++i) { r[i] = lo[i]; r[i + 8] = hi[i]; }
    return r;
}

static __device__ inline v8f wmma_bf16(v16bf a, v16bf b, v8f c) {
    // (neg_a, A, neg_b, B, c_mod, C, reuse_a, reuse_b)
    return __builtin_amdgcn_wmma_f32_16x16x32_bf16(false, a, false, b, (short)0, c,
                                                   false, false);
}

// Async DMA of 16 bytes global -> LDS (GV addressing, per-lane LDS dest VGPR).
// Tracked by ASYNCcnt; completion enforced with s_wait_asynccnt.
static __device__ inline void async_g2l_16(const void* lds_dst, const void* gsrc) {
    unsigned           lds = (unsigned)(size_t)lds_dst;       // LDS offset = addr[31:0]
    unsigned long long ga  = (unsigned long long)(size_t)gsrc;
    asm volatile("global_load_async_to_lds_b128 %0, %1, off"
                 :: "v"(lds), "v"(ga)
                 : "memory");
}

// ---------------------------------------------------------------------------
// Kernel 1: cast Q and K activations fp32 -> bf16 (row major [B,N,D])
// ---------------------------------------------------------------------------
__global__ void mab_cast_qk(const float* __restrict__ Q, const float* __restrict__ K,
                            __bf16* __restrict__ Qb, __bf16* __restrict__ Kb) {
    int i = blockIdx.x * blockDim.x + threadIdx.x;
    if (i < ACT) {
        Qb[i] = (__bf16)Q[i];
        Kb[i] = (__bf16)K[i];
    }
}

// ---------------------------------------------------------------------------
// Kernel 2: transpose-cast the four 512x512 weights.  Wt[n][k] = W[k][n]
// so WMMA B-operand columns are K-contiguous.
// ---------------------------------------------------------------------------
__global__ void mab_cast_w(const float* __restrict__ Wq, const float* __restrict__ Wk,
                           const float* __restrict__ Wv, const float* __restrict__ Wo,
                           __bf16* __restrict__ Wqt, __bf16* __restrict__ Wkt,
                           __bf16* __restrict__ Wvt, __bf16* __restrict__ Wot) {
    int i = blockIdx.x * blockDim.x + threadIdx.x;   // i = k*512 + n
    if (i < DIM * DIM) {
        int k = i >> 9, n = i & (DIM - 1);
        int o = n * DIM + k;
        Wqt[o] = (__bf16)Wq[i];
        Wkt[o] = (__bf16)Wk[i];
        Wvt[o] = (__bf16)Wv[i];
        Wot[o] = (__bf16)Wo[i];
    }
}

// ---------------------------------------------------------------------------
// Kernel 3: generic GEMM  P = A[bf16, MxK] @ Wt[bf16, NxK]^T + bias
// One wave computes a 16(M) x 64(N) tile; block = 8 waves -> 128 x 64 tile.
// mode 0: outF = P, outB = bf16(P * bscale)                   (Q projection)
// mode 1: outB = bf16(P)                                      (K projection)
// mode 2: outT[((b*H+h)*64+d)*1024 + q] = bf16(P)             (V, transposed)
// mode 3: outF = resid + relu(P)                              (output proj)
// ---------------------------------------------------------------------------
__global__ void mab_gemm(const __bf16* __restrict__ A, const __bf16* __restrict__ Wt,
                         const float* __restrict__ bias,
                         float* __restrict__ outF, __bf16* __restrict__ outB,
                         __bf16* __restrict__ outT, const float* __restrict__ resid,
                         float bscale, int mode) {
    const int lane = threadIdx.x & 31;
    const int wave = threadIdx.x >> 5;
    const int l15  = lane & 15;
    const int hi8  = (lane < 16) ? 0 : 8;    // A-operand K interleave
    const int k16  = (lane < 16) ? 0 : 16;   // unused; B also uses the hi8 interleave

    const int m0 = (blockIdx.x * 8 + wave) * 16;
    const int n0 = blockIdx.y * 64;

    v8f acc[4];
#pragma unroll
    for (int t = 0; t < 4; ++t) acc[t] = (v8f)(0.0f);

    const __bf16* arow = A + (size_t)(m0 + l15) * DIM;
    for (int k0 = 0; k0 < DIM; k0 += 32) {
        const __bf16* ap = arow + k0 + hi8;
        v16bf av = ld_pair(ap, ap + 16);
#pragma unroll
        for (int t = 0; t < 4; ++t) {
            const __bf16* bp = Wt + (size_t)(n0 + t * 16 + l15) * DIM + k0 + hi8;
            v16bf bv = ld_pair(bp, bp + 16);
            acc[t] = wmma_bf16(av, bv, acc[t]);
        }
    }

#pragma unroll
    for (int t = 0; t < 4; ++t) {
        int col = n0 + t * 16 + l15;
        float bv = bias[col];
#pragma unroll
        for (int r = 0; r < 8; ++r) {
            int row = m0 + r + hi8;                   // global M index (b*1024+q)
            float val = acc[t][r] + bv;
            size_t idx = (size_t)row * DIM + col;
            if (mode == 0) {
                outF[idx] = val;
                outB[idx] = (__bf16)(val * bscale);
            } else if (mode == 1) {
                outB[idx] = (__bf16)val;
            } else if (mode == 2) {
                int b = row >> 10, q = row & (SEQ - 1);
                int h = col >> 6, d = col & (HDIM - 1);
                outT[((size_t)(b * HEADS + h) * HDIM + d) * SEQ + q] = (__bf16)val;
            } else {
                outF[idx] = resid[idx] + fmaxf(val, 0.0f);
            }
        }
    }
}

// ---------------------------------------------------------------------------
// Kernel 4: attention with online softmax.
// Qs   : projected Q, bf16, pre-scaled by 1/sqrt(512)   [B,N,D]
// Kpb  : projected K, bf16                              [B,N,D]
// Vt   : projected V, bf16 transposed                   [B,H,hd,N]
// Qp   : projected Q, fp32 (residual)                   [B,N,D]
// Oh   : output = Qh + softmax(Q K^T) V, fp32           [B,N,D]
// One wave owns 16 q-rows x hd=64; block = 4 waves share one (b,h) and
// cooperatively stage K/V tiles into LDS with async-to-LDS DMA.
// ---------------------------------------------------------------------------
#define KP 72   // padded K-tile row length (bf16): 144 B rows, 16B aligned
#define VP 40   // padded V-tile row length (bf16):  80 B rows, 16B aligned

__global__ void mab_attn(const __bf16* __restrict__ Qs, const __bf16* __restrict__ Kpb,
                         const __bf16* __restrict__ Vt, const float* __restrict__ Qp,
                         float* __restrict__ Oh) {
    const int tid  = threadIdx.x;          // 128
    const int lane = tid & 31;
    const int wave = tid >> 5;
    const int l15  = lane & 15;
    const int hi8  = (lane < 16) ? 0 : 8;
    const int k16  = (lane < 16) ? 0 : 16;

    const int bh = blockIdx.y;
    const int b  = bh >> 3;
    const int h  = bh & (HEADS - 1);
    const int q0 = (blockIdx.x * 4 + wave) * 16;

    __shared__ __bf16 Ktile[32][KP];       // keys x hd      (rows of 64 used)
    __shared__ __bf16 Vtile[HDIM][VP];     // hd   x keys    (rows of 32 used)
    __shared__ __bf16 Pl[4][16][32];       // per-wave probs staging (C -> A layout)

    // Staging assignment: each thread copies 32 B of K and 32 B of V per step.
    const int kr = (tid * 16) >> 6, kc = (tid * 16) & 63;   // K tile [32][64]
    const int vr = (tid * 16) >> 5, vc = (tid * 16) & 31;   // V tile [64][32]
    const __bf16* ksrc0 = Kpb + ((size_t)(b * SEQ) + kr) * DIM + h * HDIM + kc;
    const __bf16* vsrc0 = Vt + ((size_t)bh * HDIM + vr) * SEQ + vc;

    // Preload this wave's scaled-Q A-operands for both hd 32-chunks.
    const __bf16* qrow = Qs + ((size_t)(b * SEQ) + q0 + l15) * DIM + h * HDIM;
    v16bf qa[2];
#pragma unroll
    for (int s = 0; s < 2; ++s) {
        const __bf16* p = qrow + s * 32 + hi8;
        qa[s] = ld_pair(p, p + 16);
    }

    v8f acc[4];
#pragma unroll
    for (int t = 0; t < 4; ++t) acc[t] = (v8f)(0.0f);
    float mrow[8], lrow[8];
#pragma unroll
    for (int r = 0; r < 8; ++r) { mrow[r] = -1e30f; lrow[r] = 0.0f; }

    for (int kb = 0; kb < SEQ; kb += 32) {
        // ---- async DMA stage of K (32x64) and V^T (64x32) tiles ----
        async_g2l_16(&Ktile[kr][kc],     ksrc0 + (size_t)kb * DIM);
        async_g2l_16(&Ktile[kr][kc + 8], ksrc0 + (size_t)kb * DIM + 8);
        async_g2l_16(&Vtile[vr][vc],     vsrc0 + kb);
        async_g2l_16(&Vtile[vr][vc + 8], vsrc0 + kb + 8);
        asm volatile("s_wait_asynccnt 0" ::: "memory");
        __syncthreads();

        // ---- scores: two 16x16 tiles (keys kb..kb+15, kb+16..kb+31) ----
        v8f S[2];
#pragma unroll
        for (int t = 0; t < 2; ++t) {
            S[t] = (v8f)(0.0f);
#pragma unroll
            for (int s = 0; s < 2; ++s) {
                const __bf16* kp = &Ktile[t * 16 + l15][s * 32 + hi8];
                v16bf kv = ld_pair(kp, kp + 16);
                S[t] = wmma_bf16(qa[s], kv, S[t]);
            }
        }

        // ---- online softmax (row stats live within 16-lane halves) ----
#pragma unroll
        for (int r = 0; r < 8; ++r) {
            float v = fmaxf(S[0][r], S[1][r]);
            for (int off = 1; off < 16; off <<= 1)
                v = fmaxf(v, __shfl_xor(v, off, 32));
            float mnew = fmaxf(mrow[r], v);
            float p0 = __expf(S[0][r] - mnew);
            float p1 = __expf(S[1][r] - mnew);
            S[0][r] = p0; S[1][r] = p1;
            float s = p0 + p1;
            for (int off = 1; off < 16; off <<= 1)
                s += __shfl_xor(s, off, 32);
            float sc = __expf(mrow[r] - mnew);
            lrow[r] = lrow[r] * sc + s;
            mrow[r] = mnew;
            acc[0][r] *= sc; acc[1][r] *= sc; acc[2][r] *= sc; acc[3][r] *= sc;
        }

        // ---- stage probs through LDS: C-layout -> A-layout bf16 16x32 ----
#pragma unroll
        for (int t = 0; t < 2; ++t)
#pragma unroll
            for (int r = 0; r < 8; ++r)
                Pl[wave][r + hi8][t * 16 + l15] = (__bf16)S[t][r];
        asm volatile("s_wait_dscnt 0" ::: "memory");
        const __bf16* pp = &Pl[wave][l15][hi8];
        v16bf pa = ld_pair(pp, pp + 16);

        // ---- acc += P @ V  (V tile rows are K-contiguous) ----
#pragma unroll
        for (int t = 0; t < 4; ++t) {
            const __bf16* vp = &Vtile[t * 16 + l15][hi8];
            v16bf vv = ld_pair(vp, vp + 16);
            acc[t] = wmma_bf16(pa, vv, acc[t]);
        }
        __syncthreads();   // protect tiles before next stage overwrites
    }

    // ---- normalize, add projected-Q residual, store fp32 ----
#pragma unroll
    for (int r = 0; r < 8; ++r) {
        float inv = 1.0f / lrow[r];
        int row = q0 + r + hi8;
#pragma unroll
        for (int t = 0; t < 4; ++t) {
            int d = t * 16 + l15;
            size_t idx = ((size_t)(b * SEQ) + row) * DIM + h * HDIM + d;
            Oh[idx] = acc[t][r] * inv + Qp[idx];
        }
    }
}

// ---------------------------------------------------------------------------
// Kernel 5: LayerNorm over last dim (512).  One row per block (128 threads).
// Optionally also writes a bf16 copy (A-operand for the next GEMM).
// ---------------------------------------------------------------------------
__global__ void mab_ln(const float* __restrict__ X, const float* __restrict__ g,
                       const float* __restrict__ bt, float* __restrict__ outF,
                       __bf16* __restrict__ outB) {
    const int row = blockIdx.x;
    const int tid = threadIdx.x;      // 128
    const float4 x = reinterpret_cast<const float4*>(X + (size_t)row * DIM)[tid];

    float s  = x.x + x.y + x.z + x.w;
    float s2 = x.x * x.x + x.y * x.y + x.z * x.z + x.w * x.w;
    for (int off = 1; off < 32; off <<= 1) {
        s  += __shfl_xor(s,  off, 32);
        s2 += __shfl_xor(s2, off, 32);
    }
    __shared__ float ws[4], ws2[4];
    if ((tid & 31) == 0) { ws[tid >> 5] = s; ws2[tid >> 5] = s2; }
    __syncthreads();
    s  = ws[0] + ws[1] + ws[2] + ws[3];
    s2 = ws2[0] + ws2[1] + ws2[2] + ws2[3];

    float mean = s * (1.0f / DIM);
    float var  = s2 * (1.0f / DIM) - mean * mean;
    float rstd = rsqrtf(var + LN_EPS);

    float xs[4] = {x.x, x.y, x.z, x.w};
    float4 y;
    float* yp = &y.x;
#pragma unroll
    for (int j = 0; j < 4; ++j) {
        int col = tid * 4 + j;
        float v = (xs[j] - mean) * rstd * g[col] + bt[col];
        yp[j] = v;
        if (outB) outB[(size_t)row * DIM + col] = (__bf16)v;
    }
    reinterpret_cast<float4*>(outF + (size_t)row * DIM)[tid] = y;
}

// ---------------------------------------------------------------------------
extern "C" void kernel_launch(void* const* d_in, const int* in_sizes, int n_in,
                              void* d_out, int out_size, void* d_ws, size_t ws_size,
                              hipStream_t stream) {
    (void)in_sizes; (void)n_in; (void)out_size; (void)ws_size;
    const float* Q  = (const float*)d_in[0];
    const float* K  = (const float*)d_in[1];
    const float* Wq = (const float*)d_in[2];
    const float* bq = (const float*)d_in[3];
    const float* Wk = (const float*)d_in[4];
    const float* bk = (const float*)d_in[5];
    const float* Wv = (const float*)d_in[6];
    const float* bv = (const float*)d_in[7];
    const float* Wo = (const float*)d_in[8];
    const float* bo = (const float*)d_in[9];
    const float* g0 = (const float*)d_in[10];
    const float* b0 = (const float*)d_in[11];
    const float* g1 = (const float*)d_in[12];
    const float* b1 = (const float*)d_in[13];
    float* out = (float*)d_out;

    char* ws = (char*)d_ws;
    size_t off = 0;
    auto nxt = [&](size_t bytes) {
        size_t r = off;
        off += (bytes + 255) & ~(size_t)255;
        return r;
    };
    __bf16* Qb   = (__bf16*)(ws + nxt((size_t)ACT * 2));
    __bf16* Kb   = (__bf16*)(ws + nxt((size_t)ACT * 2));
    __bf16* Wqt  = (__bf16*)(ws + nxt((size_t)DIM * DIM * 2));
    __bf16* Wkt  = (__bf16*)(ws + nxt((size_t)DIM * DIM * 2));
    __bf16* Wvt  = (__bf16*)(ws + nxt((size_t)DIM * DIM * 2));
    __bf16* Wot  = (__bf16*)(ws + nxt((size_t)DIM * DIM * 2));
    float*  Qp   = (float*) (ws + nxt((size_t)ACT * 4));   // also reused as Z
    __bf16* Qpb  = (__bf16*)(ws + nxt((size_t)ACT * 2));
    __bf16* Kpb  = (__bf16*)(ws + nxt((size_t)ACT * 2));
    __bf16* Vt   = (__bf16*)(ws + nxt((size_t)ACT * 2));
    float*  Oh   = (float*) (ws + nxt((size_t)ACT * 4));
    float*  Xf   = (float*) (ws + nxt((size_t)ACT * 4));
    __bf16* Xb   = (__bf16*)(ws + nxt((size_t)ACT * 2));

    const float score_scale = 1.0f / sqrtf((float)DIM);
    const int M = BATCH * SEQ;                       // 16384
    const dim3 gemm_grid(M / 128, DIM / 64);         // (128, 8)
    const dim3 attn_grid(SEQ / 64, BATCH * HEADS);   // (16, 128)

    // 1) casts
    mab_cast_qk<<<ACT / 256, 256, 0, stream>>>(Q, K, Qb, Kb);
    mab_cast_w<<<(DIM * DIM) / 256, 256, 0, stream>>>(Wq, Wk, Wv, Wo, Wqt, Wkt, Wvt, Wot);

    // 2) projections
    mab_gemm<<<gemm_grid, 256, 0, stream>>>(Qb, Wqt, bq, Qp, Qpb, nullptr, nullptr,
                                            score_scale, 0);
    mab_gemm<<<gemm_grid, 256, 0, stream>>>(Kb, Wkt, bk, nullptr, Kpb, nullptr, nullptr,
                                            1.0f, 1);
    mab_gemm<<<gemm_grid, 256, 0, stream>>>(Kb, Wvt, bv, nullptr, nullptr, Vt, nullptr,
                                            1.0f, 2);

    // 3) attention (+ projected-Q residual)
    mab_attn<<<attn_grid, 128, 0, stream>>>(Qpb, Kpb, Vt, Qp, Oh);

    // 4) LN0 -> Xf (fp32) + Xb (bf16)
    mab_ln<<<M, 128, 0, stream>>>(Oh, g0, b0, Xf, Xb);

    // 5) Z = Xf + relu(Xb @ Wo + bo)    (Z reuses the Qp buffer)
    mab_gemm<<<gemm_grid, 256, 0, stream>>>(Xb, Wot, bo, Qp, nullptr, nullptr, Xf,
                                            1.0f, 3);

    // 6) LN1 -> d_out
    mab_ln<<<M, 128, 0, stream>>>(Qp, g1, b1, out, nullptr);
}
